// SwitchEncoderBlockPipe_70265664962606
// MI455X (gfx1250) — matrix operand order repaired and checked by reading, not compiled, not run on hardware
//
#include <hip/hip_runtime.h>
#include <hip/hip_bf16.h>
#include <stdint.h>

// ---------------------------------------------------------------------------
// T5 encoder self-attention block for MI455X (gfx1250, wave32, WMMA).
// Matmuls: v_wmma_f32_16x16x32_bf16 with vectorized (b128) LDS fragment loads.
// Tile staging: global_load_async_to_lds_b128 (ASYNCcnt) where no transpose
// is needed; unrolled uniform loops everywhere (no exec-masked staging loops).
// ---------------------------------------------------------------------------

typedef __bf16 bf16;
typedef __attribute__((ext_vector_type(16))) __bf16 v16bf;
typedef __attribute__((ext_vector_type(8)))  __bf16 v8bf;
typedef __attribute__((ext_vector_type(8)))  float  v8f;

#define WMMA_BF16(a, b, c) \
  __builtin_amdgcn_wmma_f32_16x16x32_bf16(false, (a), false, (b), (short)0, (c), false, false)

#define CAT16(lo, hi) \
  __builtin_shufflevector((lo), (hi), 0, 1, 2, 3, 4, 5, 6, 7, 8, 9, 10, 11, 12, 13, 14, 15)

static constexpr int B_  = 16;
static constexpr int S_  = 512;
static constexpr int D_  = 768;
static constexpr int H_  = 12;
static constexpr int DK_ = 64;
static constexpr int INNER_ = H_ * DK_;         // 768
static constexpr int NROW_  = B_ * S_;          // 8192

// ---------------------------------------------------------------------------
// Async global -> LDS copy (CDNA5 Tensor/async path, 16 bytes per lane).
// LDS byte address = low 32 bits of the generic pointer (aperture mapping).
// Tracked by ASYNCcnt; drain with s_wait_asynccnt before the barrier.
// ---------------------------------------------------------------------------
__device__ __forceinline__ void async_copy_b128(void* lds_dst,
                                                const void* gsrc) {
  uint32_t l = (uint32_t)(uintptr_t)lds_dst;
  asm volatile("global_load_async_to_lds_b128 %0, %1, off"
               :
               : "v"(l), "v"(gsrc)
               : "memory");
}
__device__ __forceinline__ void wait_async0() {
  asm volatile("s_wait_asynccnt 0x0" ::: "memory");
}

// ---------------------------------------------------------------------------
// RMSNorm: h = x * rsqrt(mean(x^2)+eps) * w, output bf16.  One block per row.
// ---------------------------------------------------------------------------
__global__ __launch_bounds__(256) void rmsnorm_bf16_kernel(
    const float* __restrict__ x, const float* __restrict__ w,
    bf16* __restrict__ h) {
  __shared__ float red[8];
  const size_t row = blockIdx.x;
  const float* xr = x + row * (size_t)D_;
  float ss = 0.f;
#pragma unroll
  for (int t = 0; t < 3; ++t) {   // 768 = 3 * 256, uniform trip count
    float v = xr[threadIdx.x + t * 256];
    ss += v * v;
  }
  for (int o = 16; o > 0; o >>= 1) ss += __shfl_xor(ss, o, 32);
  if ((threadIdx.x & 31) == 0) red[threadIdx.x >> 5] = ss;
  __syncthreads();
  float tot = 0.f;
#pragma unroll
  for (int i = 0; i < 8; ++i) tot += red[i];
  const float rstd = rsqrtf(tot * (1.0f / D_) + 1e-6f);
  bf16* hr = h + row * (size_t)D_;
#pragma unroll
  for (int t = 0; t < 3; ++t) {
    int i = threadIdx.x + t * 256;
    hr[i] = (bf16)(xr[i] * rstd * w[i]);
  }
}

// ---------------------------------------------------------------------------
// fp32 -> bf16 cast
// ---------------------------------------------------------------------------
__global__ void cast_bf16_kernel(const float* __restrict__ src,
                                 bf16* __restrict__ dst, int n) {
  int i = blockIdx.x * blockDim.x + threadIdx.x;
  if (i < n) dst[i] = (bf16)src[i];
}

// ---------------------------------------------------------------------------
// Passthrough copies
// ---------------------------------------------------------------------------
__global__ void copy_f32x4_kernel(const float4* __restrict__ src,
                                  float4* __restrict__ dst, long long n4) {
  long long i = (long long)blockIdx.x * blockDim.x + threadIdx.x;
  if (i < n4) dst[i] = src[i];
}
__global__ void copy_f32_kernel(const float* __restrict__ src,
                                float* __restrict__ dst, int n) {
  int i = blockIdx.x * blockDim.x + threadIdx.x;
  if (i < n) dst[i] = src[i];
}
__global__ void copy_i32_to_f32_kernel(const int* __restrict__ src,
                                       float* __restrict__ dst, int n) {
  int i = blockIdx.x * blockDim.x + threadIdx.x;
  if (i < n) dst[i] = (float)src[i];
}

// ---------------------------------------------------------------------------
// A-fragment loader (16x32 bf16, row-major source, ld = 32 elements).
// Lane&15 = row; K {0..15} (lanes<16) or {16..31} (lanes>=16)
// = two contiguous 16-byte chunks -> two b128 LDS loads + register concat.
// ---------------------------------------------------------------------------
__device__ __forceinline__ v16bf load_A_frag32(const bf16* __restrict__ base) {
  const int lane = threadIdx.x & 31;
  const int kh = (lane >> 4) ? 16 : 0;
  const bf16* p = base + (lane & 15) * 32;
  v8bf lo = *(const v8bf*)(p + kh);
  v8bf hi = *(const v8bf*)(p + 8 + kh);
  return CAT16(lo, hi);
}

// ---------------------------------------------------------------------------
// Generic bf16 GEMM: C[MxN] = A[MxK] * B[KxN]  (row-major; M%128, N%64, K%32)
// Block = 128 threads (4 waves). Block tile 128x64, wave tile 32x64, K step 32.
// A tile staged via async b128 loads; B tile staged transposed (BsT[col][k])
// so B-fragments are contiguous 32B.
// mode 0: store bf16 into Cbf.  mode 1: store f32 Res + acc into Cf.
// ---------------------------------------------------------------------------
__global__ __launch_bounds__(128) void gemm_bf16_wmma_kernel(
    const bf16* __restrict__ A, const bf16* __restrict__ Bm,
    bf16* __restrict__ Cbf, float* __restrict__ Cf,
    const float* __restrict__ Res, int M, int N, int K, int mode) {
  __shared__ alignas(32) bf16 As[128 * 32];
  __shared__ alignas(32) bf16 BsT[64 * 32];  // [col][k]
  const int wave = threadIdx.x >> 5;
  const int lane = threadIdx.x & 31;
  const int col = lane & 15;
  const int kr = (lane >> 4) ? 16 : 0;
  const int half8 = (lane >> 4) * 8;
  const int mblk = blockIdx.x * 128;
  const int nblk = blockIdx.y * 64;

  // per-thread staging coordinates (uniform trip counts, fully unrolled)
  const int ar = threadIdx.x >> 2;            // A row group
  const int ac = (threadIdx.x & 3) * 8;       // A col
  const int br = threadIdx.x >> 3;            // B row
  const int bc = (threadIdx.x & 7) * 8;       // B col

  v8f acc0[4] = {};
  v8f acc1[4] = {};

  for (int k0 = 0; k0 < K; k0 += 32) {
    // A tile: 128x32 bf16, 4 async b128 copies per thread (no VGPR roundtrip)
#pragma unroll
    for (int t = 0; t < 4; ++t) {
      int r = ar + t * 32;
      async_copy_b128(&As[r * 32 + ac],
                      &A[(size_t)(mblk + r) * K + k0 + ac]);
    }
    // B tile: vector global read, transposed scalar LDS scatter
#pragma unroll
    for (int t = 0; t < 2; ++t) {
      int r = br + t * 16;
      v8bf bb = *(const v8bf*)&Bm[(size_t)(k0 + r) * N + nblk + bc];
#pragma unroll
      for (int e = 0; e < 8; ++e) BsT[(bc + e) * 32 + r] = bb[e];
    }
    wait_async0();
    __syncthreads();

    v16bf a0 = load_A_frag32(As + (wave * 32) * 32);
    v16bf a1 = load_A_frag32(As + (wave * 32 + 16) * 32);
#pragma unroll
    for (int n = 0; n < 4; ++n) {
      v16bf b = *(const v16bf*)&BsT[(n * 16 + col) * 32 + kr];
      acc0[n] = WMMA_BF16(a0, b, acc0[n]);
      acc1[n] = WMMA_BF16(a1, b, acc1[n]);
    }
    __syncthreads();
  }

#pragma unroll
  for (int n = 0; n < 4; ++n) {
#pragma unroll
    for (int v = 0; v < 8; ++v) {
      size_t gc = (size_t)(nblk + n * 16 + col);
      size_t r0 = (size_t)(mblk + wave * 32 + v + half8);
      size_t r1 = r0 + 16;
      size_t i0 = r0 * (size_t)N + gc;
      size_t i1 = r1 * (size_t)N + gc;
      if (mode == 0) {
        Cbf[i0] = (bf16)acc0[n][v];
        Cbf[i1] = (bf16)acc1[n][v];
      } else {
        Cf[i0] = Res[i0] + acc0[n][v];
        Cf[i1] = Res[i1] + acc1[n][v];
      }
    }
  }
}

// ---------------------------------------------------------------------------
// T5 relative position bucket (bidirectional, 32 buckets, max_distance 128)
// ---------------------------------------------------------------------------
__device__ __forceinline__ int rel_bucket(int rel) {
  const int nb = 16;         // num_buckets / 2
  const int max_exact = 8;   // nb / 2
  int ret = (rel > 0) ? nb : 0;
  int n = (rel < 0) ? -rel : rel;
  if (n < max_exact) return ret + n;
  float nf = (float)(n < 1 ? 1 : n);
  // log(n/8) / log(128/8) * 8
  int vl = max_exact + (int)(__logf(nf * 0.125f) * (1.0f / 2.7725887f) * 8.0f);
  if (vl > nb - 1) vl = nb - 1;
  return ret + vl;
}

// ---------------------------------------------------------------------------
// Flash-style attention: one block per (64-query strip, head, batch).
// 4 waves, each wave owns 16 queries. Keys processed in chunks of 32.
// K tile staged with async b128 (layout [key][dk] -> score B-frags contiguous)
// V tile staged transposed (VtT[dk][key] -> context B-frags contiguous).
// ---------------------------------------------------------------------------
__global__ __launch_bounds__(128) void attn_kernel(
    const bf16* __restrict__ qbuf, const bf16* __restrict__ kbuf,
    const bf16* __restrict__ vbuf, const float* __restrict__ posbias,
    const float* __restrict__ relemb, bf16* __restrict__ ctxbuf) {
  __shared__ alignas(32) bf16 Kt[32 * 64];     // [key][dk]
  __shared__ alignas(32) bf16 VtT[64 * 32];    // [dk][key]
  __shared__ alignas(32) bf16 Pb[4 * 16 * 32]; // per-wave P [row][key] ld=32

  const int wave = threadIdx.x >> 5;
  const int lane = threadIdx.x & 31;
  const int col = lane & 15;
  const int half8 = (lane >> 4) * 8;
  const int kh = (lane >> 4) ? 8 : 0;
  const int kr = (lane >> 4) ? 16 : 0;
  const int hIdx = blockIdx.y;
  const int bz = blockIdx.z;
  const int qbase = blockIdx.x * 64 + wave * 16;

  // per-thread staging coordinates
  const int skk = threadIdx.x >> 3;            // key row
  const int sc  = (threadIdx.x & 7) * 8;       // dk col

  // Q fragments in A layout, K-dim = dk (64 -> two 32-chunks), global v8bf
  const bf16* qp =
      qbuf + ((size_t)(bz * S_ + qbase + col)) * INNER_ + hIdx * DK_;
  v16bf qa0, qa1;
  {
    v8bf l0 = *(const v8bf*)(qp + kr);
    v8bf h0 = *(const v8bf*)(qp + 8 + kr);
    v8bf l1 = *(const v8bf*)(qp + 32 + kr);
    v8bf h1 = *(const v8bf*)(qp + 40 + kr);
    qa0 = CAT16(l0, h0);
    qa1 = CAT16(l1, h1);
  }

  float m[8], l[8];
#pragma unroll
  for (int v = 0; v < 8; ++v) { m[v] = -3.0e38f; l[v] = 0.f; }
  v8f acc[4] = {};

  bf16* Pw = Pb + wave * 512;

  for (int kc = 0; kc < S_; kc += 32) {
    // stage K (async, row-major) and V (transposed via VGPRs)
#pragma unroll
    for (int t = 0; t < 2; ++t) {
      int kk = skk + t * 16;
      size_t src = ((size_t)(bz * S_ + kc + kk)) * INNER_ + hIdx * DK_ + sc;
      async_copy_b128(&Kt[kk * 64 + sc], &kbuf[src]);
      v8bf vv = *(const v8bf*)&vbuf[src];
#pragma unroll
      for (int e = 0; e < 8; ++e) VtT[(sc + e) * 32 + kk] = vv[e];
    }
    wait_async0();
    __syncthreads();

    // scores: two 16x16 tiles (keys kc..kc+15, kc+16..kc+31), K-dim = dk
    v8f s0 = {}, s1 = {};
    {
      v16bf b;
      b = *(const v16bf*)&Kt[col * 64 + kr];
      s0 = WMMA_BF16(qa0, b, s0);
      b = *(const v16bf*)&Kt[col * 64 + 32 + kr];
      s0 = WMMA_BF16(qa1, b, s0);
      b = *(const v16bf*)&Kt[(16 + col) * 64 + kr];
      s1 = WMMA_BF16(qa0, b, s1);
      b = *(const v16bf*)&Kt[(16 + col) * 64 + 32 + kr];
      s1 = WMMA_BF16(qa1, b, s1);
    }

    // bias: relative-position embedding + passed-in mask bias
    {
      const int q0 = qbase + half8;
      const int k0i = kc + col;
      const int k1i = kc + 16 + col;
#pragma unroll
      for (int v = 0; v < 8; ++v) {
        int q = q0 + v;
        size_t pb = (((size_t)bz * H_ + hIdx) * S_ + q) * (size_t)S_;
        s0[v] += relemb[rel_bucket(k0i - q) * H_ + hIdx] + posbias[pb + k0i];
        s1[v] += relemb[rel_bucket(k1i - q) * H_ + hIdx] + posbias[pb + k1i];
      }
    }

    // online softmax (rows live across 16-lane halves; xor 1/2/4/8 reduces)
    float p0v[8], p1v[8];
#pragma unroll
    for (int v = 0; v < 8; ++v) {
      float cm = fmaxf(s0[v], s1[v]);
#pragma unroll
      for (int o = 1; o < 16; o <<= 1) cm = fmaxf(cm, __shfl_xor(cm, o, 32));
      float mn = fmaxf(m[v], cm);
      float sc2 = __expf(m[v] - mn);
      float p0 = __expf(s0[v] - mn);
      float p1 = __expf(s1[v] - mn);
      float rs = p0 + p1;
#pragma unroll
      for (int o = 1; o < 16; o <<= 1) rs += __shfl_xor(rs, o, 32);
      l[v] = l[v] * sc2 + rs;
      m[v] = mn;
      p0v[v] = p0;
      p1v[v] = p1;
#pragma unroll
      for (int n = 0; n < 4; ++n) acc[n][v] *= sc2;
    }

    // stage P (C layout -> row-major LDS) and reload as A fragment
#pragma unroll
    for (int v = 0; v < 8; ++v) {
      Pw[(v + half8) * 32 + col]      = (bf16)p0v[v];
      Pw[(v + half8) * 32 + 16 + col] = (bf16)p1v[v];
    }
    v16bf pa = load_A_frag32(Pw);

    // ctx += P (16x32) @ V (32 keys x 64 dk), four 16-wide N tiles
#pragma unroll
    for (int n = 0; n < 4; ++n) {
      v16bf b = *(const v16bf*)&VtT[(n * 16 + col) * 32 + kr];
      acc[n] = WMMA_BF16(pa, b, acc[n]);
    }
    __syncthreads();
  }

  // normalize and store ctx as bf16 for the output projection GEMM
#pragma unroll
  for (int n = 0; n < 4; ++n) {
#pragma unroll
    for (int v = 0; v < 8; ++v) {
      int q = qbase + v + half8;
      size_t idx =
          ((size_t)(bz * S_ + q)) * INNER_ + hIdx * DK_ + n * 16 + col;
      ctxbuf[idx] = (bf16)(acc[n][v] / l[v]);
    }
  }
}

// ---------------------------------------------------------------------------
// Host-side launcher
// ---------------------------------------------------------------------------
extern "C" void kernel_launch(void* const* d_in, const int* in_sizes, int n_in,
                              void* d_out, int out_size, void* d_ws,
                              size_t ws_size, hipStream_t stream) {
  const float* x       = (const float*)d_in[0];   // [B,S,D]
  const float* posbias = (const float*)d_in[1];   // [B,H,S,S]
  const float* extmask = (const float*)d_in[2];   // [B,1,1,S]
  const int*   ids     = (const int*)d_in[3];     // [B,S]
  const float* decmask = (const float*)d_in[4];   // [B,S]
  const float* encmask = (const float*)d_in[5];   // [B,S]
  const float* lnw     = (const float*)d_in[6];   // [D]
  const float* wq      = (const float*)d_in[7];   // [D,768]
  const float* wk      = (const float*)d_in[8];
  const float* wv      = (const float*)d_in[9];
  const float* wo      = (const float*)d_in[10];  // [768,D]
  const float* relemb  = (const float*)d_in[11];  // [32,H]

  // workspace layout (~68 MB total, 256B aligned)
  char* ws = (char*)d_ws;
  size_t off = 0;
  auto alloc = [&](size_t bytes) -> void* {
    void* p = ws + off;
    off = (off + bytes + 255) & ~(size_t)255;
    return p;
  };
  const size_t act_bytes = (size_t)NROW_ * INNER_ * sizeof(bf16);
  const size_t w_bytes   = (size_t)D_ * INNER_ * sizeof(bf16);
  bf16* hb   = (bf16*)alloc(act_bytes);
  bf16* wqb  = (bf16*)alloc(w_bytes);
  bf16* wkb  = (bf16*)alloc(w_bytes);
  bf16* wvb  = (bf16*)alloc(w_bytes);
  bf16* wob  = (bf16*)alloc(w_bytes);
  bf16* qb   = (bf16*)alloc(act_bytes);
  bf16* kb   = (bf16*)alloc(act_bytes);
  bf16* vb   = (bf16*)alloc(act_bytes);
  bf16* ctxb = (bf16*)alloc(act_bytes);

  // 1. RMSNorm -> bf16 h
  rmsnorm_bf16_kernel<<<NROW_, 256, 0, stream>>>(x, lnw, hb);

  // 2. weights -> bf16
  const int wn = D_ * INNER_;
  cast_bf16_kernel<<<(wn + 255) / 256, 256, 0, stream>>>(wq, wqb, wn);
  cast_bf16_kernel<<<(wn + 255) / 256, 256, 0, stream>>>(wk, wkb, wn);
  cast_bf16_kernel<<<(wn + 255) / 256, 256, 0, stream>>>(wv, wvb, wn);
  cast_bf16_kernel<<<(wn + 255) / 256, 256, 0, stream>>>(wo, wob, wn);

  // 3. Q/K/V projections (8192 x 768 x 768), bf16 outputs
  dim3 ggrid(NROW_ / 128, INNER_ / 64);
  gemm_bf16_wmma_kernel<<<ggrid, 128, 0, stream>>>(hb, wqb, qb, nullptr,
                                                   nullptr, NROW_, INNER_, D_, 0);
  gemm_bf16_wmma_kernel<<<ggrid, 128, 0, stream>>>(hb, wkb, kb, nullptr,
                                                   nullptr, NROW_, INNER_, D_, 0);
  gemm_bf16_wmma_kernel<<<ggrid, 128, 0, stream>>>(hb, wvb, vb, nullptr,
                                                   nullptr, NROW_, INNER_, D_, 0);

  // 4. attention (flash-style, WMMA scores + WMMA context)
  attn_kernel<<<dim3(S_ / 64, H_, B_), 128, 0, stream>>>(qb, kb, vb, posbias,
                                                         relemb, ctxb);

  // 5. output projection + residual, f32 into d_out[0]
  float* out0 = (float*)d_out;
  gemm_bf16_wmma_kernel<<<ggrid, 128, 0, stream>>>(ctxb, wob, nullptr, out0, x,
                                                   NROW_, D_, INNER_, 1);

  // 6. tuple passthrough outputs, concatenated flat after out0
  const long long n_bias = (long long)B_ * H_ * S_ * S_;  // 50331648
  const int n_small = B_ * S_;                            // 8192
  float* o1 = out0 + (size_t)NROW_ * D_;
  copy_f32x4_kernel<<<(unsigned)((n_bias / 4 + 255) / 256), 256, 0, stream>>>(
      (const float4*)posbias, (float4*)o1, n_bias / 4);
  float* o2 = o1 + n_bias;
  copy_f32_kernel<<<(n_small + 255) / 256, 256, 0, stream>>>(extmask, o2,
                                                             n_small);
  float* o3 = o2 + n_small;
  copy_i32_to_f32_kernel<<<(n_small + 255) / 256, 256, 0, stream>>>(ids, o3,
                                                                    n_small);
  float* o4 = o3 + n_small;
  copy_f32_kernel<<<(n_small + 255) / 256, 256, 0, stream>>>(decmask, o4,
                                                             n_small);
  float* o5 = o4 + n_small;
  copy_f32_kernel<<<(n_small + 255) / 256, 256, 0, stream>>>(encmask, o5,
                                                             n_small);
}
